// GAT_encoder_11106785427983
// MI455X (gfx1250) — matrix operand order repaired and checked by reading, not compiled, not run on hardware
//
#include <hip/hip_runtime.h>
#include <hip/hip_bf16.h>
#include <math.h>

// ---------------------------------------------------------------------------
// GAT encoder for MI455X (gfx1250, wave32).
//  - Dense per-head projections run as bf16 WMMA GEMMs (v_wmma_f32_16x16x32_bf16)
//    with block-cooperative bf16 LDS staging and a fragment-layout pre-packed B.
//  - Edge softmax/scatter phase is the roofline bottleneck (~820MB gather +
//    205M f32 atomics per layer); kept coalesced so atomics resolve in L2,
//    with global_prefetch to hide gather latency.
// Sizes (reference): N=50000, E=800000, F_IN=512, HID=64, OUT=32, H=4.
// ---------------------------------------------------------------------------

typedef __attribute__((ext_vector_type(16))) __bf16 v16bf;
typedef __attribute__((ext_vector_type(8)))  __bf16 v8bf;
typedef __attribute__((ext_vector_type(4)))  __bf16 v4bf;
typedef __attribute__((ext_vector_type(8)))  float  v8f;

#define GAT_N   50000
#define GAT_E   800000
#define GAT_H   4

// float -> bf16, round-to-nearest-even (integer ops only; always lowers)
static __device__ __forceinline__ __bf16 f2bf(float f) {
    unsigned u = __float_as_uint(f);
    u += 0x7fffu + ((u >> 16) & 1u);
    unsigned short hs = (unsigned short)(u >> 16);
    return __builtin_bit_cast(__bf16, hs);
}

// ordered-uint encoding so unsigned atomicMax == float max
static __device__ __forceinline__ unsigned f2ord(float f) {
    unsigned u = __float_as_uint(f);
    return (u & 0x80000000u) ? ~u : (u | 0x80000000u);
}
static __device__ __forceinline__ float ord2f(unsigned o) {
    return __uint_as_float((o & 0x80000000u) ? (o ^ 0x80000000u) : ~o);
}

// ---------------------------------------------------------------------------
// Pack W (H, K, O) into B-fragment-friendly bf16 layout:
//   Bp[((k/32)*Ncols + c)*32 + (k%32)] = bf16( W[h=c/O][k][o=c%O] )
// so a lane's B fragment (16 consecutive K at half-wave base) is one
// contiguous 32-byte load.
// ---------------------------------------------------------------------------
__global__ void pack_b(const float* __restrict__ W, __bf16* __restrict__ Bp,
                       int K, int Ncols, int O)
{
    int t = blockIdx.x * blockDim.x + threadIdx.x;
    if (t >= K * Ncols) return;
    int c = t / K, k = t - c * K;           // k fastest -> Bp writes contiguous
    int h = c / O, o = c - h * O;
    Bp[((long)(k >> 5) * Ncols + c) * 32 + (k & 31)] =
        f2bf(W[(long)h * K * O + (long)k * O + o]);
}

// ---------------------------------------------------------------------------
// WMMA GEMM:  C[M x Ncols] = (ReLU?)A[M x K] @ B[K x Ncols], bf16 inputs.
// Block = 256 threads = 8 waves; covers 2 m-tiles (32 rows) x all Ncols.
// Per m-tile: 4 waves, each owning NTW n-tiles (NTW*64 == Ncols/... Ncols==64*NTW).
// A 2x16x32 tile staged cooperatively into LDS as bf16 each k-step; each wave's
// A fragment is two ds_load_b128 and feeds NTW consecutive v_wmma ops.
// Requires M%16==0, K%32==0 (holds: M=50000=3125*16).
// ---------------------------------------------------------------------------
template <int RELU, int NTW>
__global__ __launch_bounds__(256) void gemm_wmma_bf16(
        const float* __restrict__ A, const __bf16* __restrict__ Bp,
        float* __restrict__ C, int M, int K, int Ncols)
{
    __shared__ __bf16 at[2][16][40];        // row stride 40 bf16 = 80B (16B-aligned)

    const int tid       = threadIdx.x;
    const int lane      = tid & 31;
    const int wave      = tid >> 5;
    const int tl        = wave >> 2;        // which of the 2 m-tiles
    const int ntBase    = (wave & 3) * NTW; // first n-tile owned by this wave

    const int Mtiles = M >> 4;
    const int mt     = blockIdx.x * 2 + tl;
    const bool active = (mt < Mtiles);      // wave-uniform
    const int m0     = mt << 4;

    // staging decomposition: 256 float4 chunks = 2 tiles x 16 rows x 8 chunks
    const int sTile = tid >> 7;
    const int sRow  = (tid >> 3) & 15;
    const int sC4   = (tid & 7) << 2;
    int gRow = blockIdx.x * 32 + sTile * 16 + sRow;
    if (gRow > M - 1) gRow = M - 1;         // clamp (data unused by inactive tile)

    const int kbA = (lane >> 4) * 16;       // A-frag K base (bf16 idx) per half-wave
    const int kbB = (lane >> 4) * 16;       // B-frag K base per half-wave
    const int aRow = lane & 15;

    v8f acc[NTW];
    #pragma unroll
    for (int j = 0; j < NTW; ++j) acc[j] = (v8f){};

    for (int k0 = 0; k0 < K; k0 += 32) {
        // ---- cooperative stage: global f32 -> LDS bf16 ----
        {
            const float4 v = *(const float4*)(A + (long)gRow * K + k0 + sC4);
            float x0 = v.x, x1 = v.y, x2 = v.z, x3 = v.w;
            if (RELU) {
                x0 = fmaxf(x0, 0.f); x1 = fmaxf(x1, 0.f);
                x2 = fmaxf(x2, 0.f); x3 = fmaxf(x3, 0.f);
            }
            v4bf p = { f2bf(x0), f2bf(x1), f2bf(x2), f2bf(x3) };
            *(v4bf*)&at[sTile][sRow][sC4] = p;
        }
        __syncthreads();

        if (active) {
            // ---- A fragment: two contiguous 16B LDS reads ----
            v8bf alo = *(const v8bf*)&at[tl][aRow][kbA];
            v8bf ahi = *(const v8bf*)&at[tl][aRow][8 + kbA];
            v16bf afrag = __builtin_shufflevector(alo, ahi,
                    0,1,2,3,4,5,6,7,8,9,10,11,12,13,14,15);

            const int ks = k0 >> 5;
            #pragma unroll
            for (int j = 0; j < NTW; ++j) {
                int c = (ntBase + j) * 16 + aRow;
                v16bf bfrag = *(const v16bf*)(Bp + ((long)ks * Ncols + c) * 32 + kbB);
                acc[j] = __builtin_amdgcn_wmma_f32_16x16x32_bf16(
                             false, afrag, false, bfrag, (short)0, acc[j],
                             false, false);
            }
        }
        __syncthreads();
    }

    if (active) {
        // C layout: lane holds column (lane&15); VGPR r holds row r + 8*(lane>>4)
        #pragma unroll
        for (int j = 0; j < NTW; ++j) {
            const int colBase = (ntBase + j) << 4;
            #pragma unroll
            for (int r = 0; r < 8; ++r) {
                int mm = m0 + r + ((lane >> 4) << 3);
                C[(long)mm * Ncols + colBase + aRow] = acc[j][r];
            }
        }
    }
}

// ---------------------------------------------------------------------------
// s[n,h] = <Wh[n,h,:], a[h,0:O]> ;  d[n,h] = <Wh[n,h,:], a[h,O:2O]>
// ---------------------------------------------------------------------------
__global__ void sd_kernel(const float* __restrict__ Wh, const float* __restrict__ a,
                          float* __restrict__ s, float* __restrict__ d, int N, int O)
{
    int t = blockIdx.x * blockDim.x + threadIdx.x;
    if (t >= N * GAT_H) return;
    int n = t >> 2, h = t & 3;
    const float* w  = Wh + (long)n * GAT_H * O + h * O;
    const float* as = a + h * 2 * O;
    const float* ad = as + O;
    float ss = 0.f, dd = 0.f;
    for (int o = 0; o < O; ++o) {
        float v = w[o];
        ss = fmaf(v, as[o], ss);
        dd = fmaf(v, ad[o], dd);
    }
    s[t] = ss; d[t] = dd;
}

// ---------------------------------------------------------------------------
// Pass 1: edge_e = -leaky_relu(s[src]+d[dst]); track global per-head max
// ---------------------------------------------------------------------------
__global__ void edge_logits(const float* __restrict__ s, const float* __restrict__ d,
                            const int* __restrict__ src, const int* __restrict__ dst,
                            float* __restrict__ ee, unsigned* __restrict__ emax, int E)
{
    __shared__ unsigned sm[GAT_H];
    int e = blockIdx.x * blockDim.x + threadIdx.x;
    if (threadIdx.x < GAT_H) sm[threadIdx.x] = 0u;
    __syncthreads();
    if (e < E) {
        int si = src[e], di = dst[e];
        #pragma unroll
        for (int h = 0; h < GAT_H; ++h) {
            float ev = s[si * GAT_H + h] + d[di * GAT_H + h];
            float lr = ev > 0.f ? ev : 0.2f * ev;      // leaky_relu, slope 0.2
            float v  = -lr;
            ee[e * GAT_H + h] = v;
            atomicMax(&sm[h], f2ord(v));
        }
    }
    __syncthreads();
    if (threadIdx.x < GAT_H) atomicMax(&emax[threadIdx.x], sm[threadIdx.x]);
}

// ---------------------------------------------------------------------------
// Pass 2: ee <- exp(ee - max);  denom[src,h] += ee
// ---------------------------------------------------------------------------
__global__ void edge_exp(float* __restrict__ ee, const unsigned* __restrict__ emax,
                         const int* __restrict__ src, float* __restrict__ denom, int E)
{
    int t = blockIdx.x * blockDim.x + threadIdx.x;
    if (t >= E * GAT_H) return;
    int e = t >> 2, h = t & 3;
    float ex = __expf(ee[t] - ord2f(emax[h]));
    ee[t] = ex;
    atomicAdd(&denom[src[e] * GAT_H + h], ex);
}

// ---------------------------------------------------------------------------
// Pass 2.5: ee <- ee / (denom[src] + eps)   (hoists E*H divisions out of scatter)
// ---------------------------------------------------------------------------
__global__ void edge_att(float* __restrict__ ee, const float* __restrict__ denom,
                         const int* __restrict__ src, int E)
{
    int t = blockIdx.x * blockDim.x + threadIdx.x;
    if (t >= E * GAT_H) return;
    int e = t >> 2, h = t & 3;
    ee[t] = ee[t] / (denom[src[e] * GAT_H + h] + 1e-10f);
}

// ---------------------------------------------------------------------------
// Pass 3: hout[src,h,o] += att[e,h] * Wh[dst,h,o]
// One thread per (edge, o); consecutive threads share the edge -> coalesced
// gathers and coalesced atomics. Prefetch the gather row a few edges ahead.
// ---------------------------------------------------------------------------
template <int O>
__global__ void edge_scatter(const float* __restrict__ att,
                             const int* __restrict__ src, const int* __restrict__ dst,
                             const float* __restrict__ Wh, float* __restrict__ hout, int E)
{
    int t = blockIdx.x * blockDim.x + threadIdx.x;
    if (t >= E * O) return;
    int e = t / O;
    int o = t - e * O;
    int si = src[e], di = dst[e];

    // pull a future edge's Wh row toward the caches (global_prefetch_b8)
    int ep = e + 32;
    if (ep < E && o == 0) {
        __builtin_prefetch(&Wh[(long)dst[ep] * (GAT_H * O)], 0, 1);
        __builtin_prefetch(&hout[(long)src[ep] * (GAT_H * O)], 1, 1);
    }

    #pragma unroll
    for (int h = 0; h < GAT_H; ++h) {
        float a = att[e * GAT_H + h];
        atomicAdd(&hout[(long)si * (GAT_H * O) + h * O + o],
                  a * Wh[(long)di * (GAT_H * O) + h * O + o]);
    }
}

// ---------------------------------------------------------------------------
// Final: out[n,o] = relu(mean_h h2[n,h,o])
// ---------------------------------------------------------------------------
__global__ void mean_relu(const float* __restrict__ h2, float* __restrict__ out, int N)
{
    int t = blockIdx.x * blockDim.x + threadIdx.x;
    if (t >= N * 32) return;
    int n = t >> 5, o = t & 31;
    float acc = 0.f;
    #pragma unroll
    for (int h = 0; h < GAT_H; ++h) acc += h2[(long)n * 128 + h * 32 + o];
    out[t] = fmaxf(acc * 0.25f, 0.f);
}

// ---------------------------------------------------------------------------
extern "C" void kernel_launch(void* const* d_in, const int* in_sizes, int n_in,
                              void* d_out, int out_size, void* d_ws, size_t ws_size,
                              hipStream_t stream)
{
    (void)in_sizes; (void)n_in; (void)out_size; (void)ws_size;

    const float* features = (const float*)d_in[0];   // N x 512
    const int*   src      = (const int*)  d_in[1];   // E
    const int*   dst      = (const int*)  d_in[2];   // E
    const float* W1       = (const float*)d_in[3];   // 4 x 512 x 64
    const float* a1       = (const float*)d_in[4];   // 4 x 128
    const float* W2       = (const float*)d_in[5];   // 4 x 256 x 32
    const float* a2       = (const float*)d_in[6];   // 4 x 64
    float*       out      = (float*)d_out;           // N x 32

    const int N = GAT_N, E = GAT_E;

    // workspace carve-up (floats); total ~35.7M floats (~143 MB)
    float* ws   = (float*)d_ws;
    float* wh1  = ws;                      // N*256
    float* h1   = wh1  + (long)N * 256;    // N*256
    float* ee   = h1   + (long)N * 256;    // E*4
    float* sbuf = ee   + (long)E * 4;      // N*4
    float* dbuf = sbuf + (long)N * 4;      // N*4
    float* den  = dbuf + (long)N * 4;      // N*4
    float* h2   = den  + (long)N * 4;      // N*128
    __bf16* bp  = (__bf16*)(h2 + (long)N * 128);       // 512*256 bf16 (reused)
    unsigned* emax = (unsigned*)((float*)bp + 65536);  // 4
    float* wh2  = wh1;                     // alias: Wh1 dead once h1 is built

    const int TPB = 256;
    const int Mtiles = N / 16;             // 3125
    const int gemmBlocks = (Mtiles + 1) / 2;

    // ---------------- layer 1 ----------------
    hipMemsetAsync(h1,   0, (size_t)N * 256 * 4, stream);
    hipMemsetAsync(den,  0, (size_t)N * 4 * 4,   stream);
    hipMemsetAsync(emax, 0, GAT_H * 4,           stream);

    pack_b<<<(512 * 256 + TPB - 1) / TPB, TPB, 0, stream>>>(W1, bp, 512, 256, 64);
    gemm_wmma_bf16<0, 4><<<gemmBlocks, TPB, 0, stream>>>(features, bp, wh1, N, 512, 256);

    sd_kernel<<<(N * GAT_H + TPB - 1) / TPB, TPB, 0, stream>>>(wh1, a1, sbuf, dbuf, N, 64);
    edge_logits<<<(E + TPB - 1) / TPB, TPB, 0, stream>>>(sbuf, dbuf, src, dst, ee, emax, E);
    edge_exp<<<(E * GAT_H + TPB - 1) / TPB, TPB, 0, stream>>>(ee, emax, src, den, E);
    edge_att<<<(E * GAT_H + TPB - 1) / TPB, TPB, 0, stream>>>(ee, den, src, E);
    edge_scatter<64><<<(E * 64 + TPB - 1) / TPB, TPB, 0, stream>>>(ee, src, dst, wh1, h1, E);

    // ---------------- layer 2 ----------------
    pack_b<<<(256 * 128 + TPB - 1) / TPB, TPB, 0, stream>>>(W2, bp, 256, 128, 32);
    gemm_wmma_bf16<1, 2><<<gemmBlocks, TPB, 0, stream>>>(h1, bp, wh2, N, 256, 128);

    hipMemsetAsync(den,  0, (size_t)N * 4 * 4,   stream);
    hipMemsetAsync(emax, 0, GAT_H * 4,           stream);
    hipMemsetAsync(h2,   0, (size_t)N * 128 * 4, stream);

    sd_kernel<<<(N * GAT_H + TPB - 1) / TPB, TPB, 0, stream>>>(wh2, a2, sbuf, dbuf, N, 32);
    edge_logits<<<(E + TPB - 1) / TPB, TPB, 0, stream>>>(sbuf, dbuf, src, dst, ee, emax, E);
    edge_exp<<<(E * GAT_H + TPB - 1) / TPB, TPB, 0, stream>>>(ee, emax, src, den, E);
    edge_att<<<(E * GAT_H + TPB - 1) / TPB, TPB, 0, stream>>>(ee, den, src, E);
    edge_scatter<32><<<(E * 32 + TPB - 1) / TPB, TPB, 0, stream>>>(ee, src, dst, wh2, h2, E);

    mean_relu<<<(N * 32 + TPB - 1) / TPB, TPB, 0, stream>>>(h2, out, N);
}
